// DPPN_gg_78443282694825
// MI455X (gfx1250) — matrix operand order repaired and checked by reading, not compiled, not run on hardware
//
#include <hip/hip_runtime.h>
#include <hip/hip_bf16.h>
#include <math.h>
#include <stdint.h>

// Problem constants (fixed by the reference).
#define N_IMG   65536
#define IMG_DIM 2048
#define N_CLS   2048
#define ATT_DIM 1024
#define ATT_HC  512
#define TEMP    10.0f
#define THRESH  0.17364817766693041f   // cos(80 deg)

typedef __attribute__((ext_vector_type(16))) __bf16 v16bf;
typedef __attribute__((ext_vector_type(8)))  __bf16 v8bf;   // 16-byte packet
typedef __attribute__((ext_vector_type(4)))  __bf16 v4bf;   // 8-byte packet
typedef __attribute__((ext_vector_type(8)))  float  v8f;
typedef __attribute__((ext_vector_type(4)))  float  v4f;

// ---------------------------------------------------------------- utilities

__global__ void k_zero_f32(float* __restrict__ p, int n) {
    int i = blockIdx.x * blockDim.x + threadIdx.x;
    if (i < n) p[i] = 0.0f;
}

// 4-wide f32 -> bf16 (n must be a multiple of 4; all our sizes are)
__global__ void k_f32_to_bf16_x4(const float* __restrict__ x, __bf16* __restrict__ y, int n4) {
    int i = blockIdx.x * blockDim.x + threadIdx.x;
    if (i < n4) {
        v4f v = *(const v4f*)(x + 4 * (size_t)i);
        v4bf o;
        o[0] = (__bf16)v[0]; o[1] = (__bf16)v[1];
        o[2] = (__bf16)v[2]; o[3] = (__bf16)v[3];
        *(v4bf*)(y + 4 * (size_t)i) = o;
    }
}

// ------------------------------------------------- segment mean (protos)

// One block per image row: stream 512 MB once; fp32 atomics land in L2
// (sums is 16 MB << 192 MB global L2).
__global__ __launch_bounds__(256)
void k_scatter_add(const float* __restrict__ feats, const int* __restrict__ labels,
                   float* __restrict__ sums, float* __restrict__ counts) {
    int img = blockIdx.x;
    int lab = labels[img];
    const float* src = feats + (size_t)img * IMG_DIM;
    float* dst = sums + (size_t)lab * IMG_DIM;
    for (int j4 = threadIdx.x; j4 < IMG_DIM / 4; j4 += blockDim.x) {
        v4f v = *(const v4f*)(src + 4 * (size_t)j4);
        float* d4 = dst + 4 * (size_t)j4;
        atomicAdd(&d4[0], v[0]);
        atomicAdd(&d4[1], v[1]);
        atomicAdd(&d4[2], v[2]);
        atomicAdd(&d4[3], v[3]);
    }
    if (threadIdx.x == 0)
        atomicAdd(&counts[lab], 1.0f);
}

// protos_bf[c] = bf16( sums[tpl[c]] / counts[tpl[c]] )
__global__ __launch_bounds__(256)
void k_protos(const float* __restrict__ sums, const float* __restrict__ counts,
              const int* __restrict__ tpl, __bf16* __restrict__ protos_bf) {
    int c = blockIdx.x;
    int s = tpl[c];
    float inv = 1.0f / counts[s];
    const float* row = sums + (size_t)s * IMG_DIM;
    __bf16* dst = protos_bf + (size_t)c * IMG_DIM;
    for (int j4 = threadIdx.x; j4 < IMG_DIM / 4; j4 += blockDim.x) {
        v4f v = *(const v4f*)(row + 4 * (size_t)j4);
        v4bf o;
        o[0] = (__bf16)(v[0] * inv); o[1] = (__bf16)(v[1] * inv);
        o[2] = (__bf16)(v[2] * inv); o[3] = (__bf16)(v[3] * inv);
        *(v4bf*)(dst + 4 * (size_t)j4) = o;
    }
}

// ---------------------------------------------------- rownorm -> bf16

__global__ __launch_bounds__(256)
void k_rownorm_bf16(const float* __restrict__ x, __bf16* __restrict__ y, int cols) {
    __shared__ float red[256];
    int r = blockIdx.x;
    const float* row = x + (size_t)r * cols;
    float ss = 0.0f;
    for (int j = threadIdx.x; j < cols; j += blockDim.x) {
        float v = row[j];
        ss += v * v;
    }
    red[threadIdx.x] = ss;
    __syncthreads();
    for (int s = 128; s > 0; s >>= 1) {
        if (threadIdx.x < s) red[threadIdx.x] += red[threadIdx.x + s];
        __syncthreads();
    }
    float inv = 1.0f / fmaxf(sqrtf(red[0]), 1e-8f);
    __bf16* dst = y + (size_t)r * cols;
    for (int j = threadIdx.x; j < cols; j += blockDim.x)
        dst[j] = (__bf16)(row[j] * inv);
}

// ------------------------------- thresholded row softmax -> bf16 weights

__global__ __launch_bounds__(256)
void k_mask_softmax_bf16(const float* __restrict__ d, __bf16* __restrict__ out, int cols) {
    __shared__ float red[256];
    int r = blockIdx.x;
    const float* row = d + (size_t)r * cols;

    float mx = -3.4e38f;
    for (int j = threadIdx.x; j < cols; j += blockDim.x) {
        float v = row[j];
        float x = (v > THRESH) ? v * TEMP : -3.0e38f;
        mx = fmaxf(mx, x);
    }
    red[threadIdx.x] = mx;
    __syncthreads();
    for (int s = 128; s > 0; s >>= 1) {
        if (threadIdx.x < s) red[threadIdx.x] = fmaxf(red[threadIdx.x], red[threadIdx.x + s]);
        __syncthreads();
    }
    mx = red[0];
    __syncthreads();

    float sum = 0.0f;
    for (int j = threadIdx.x; j < cols; j += blockDim.x) {
        float v = row[j];
        float x = (v > THRESH) ? v * TEMP : -3.0e38f;
        sum += __expf(x - mx);
    }
    red[threadIdx.x] = sum;
    __syncthreads();
    for (int s = 128; s > 0; s >>= 1) {
        if (threadIdx.x < s) red[threadIdx.x] += red[threadIdx.x + s];
        __syncthreads();
    }
    float inv = 1.0f / red[0];

    __bf16* dst = out + (size_t)r * cols;
    for (int j = threadIdx.x; j < cols; j += blockDim.x) {
        float v = row[j];
        float x = (v > THRESH) ? v * TEMP : -3.0e38f;
        dst[j] = (__bf16)(__expf(x - mx) * inv);
    }
}

// ---------------------------------------------- bf16 WMMA tiled GEMM
// C[M x N](f32) = A[M x K](bf16, row-major) * B(bf16)
//   TRANSB == 0 : B is row-major [K x N]
//   TRANSB == 1 : B is row-major [N x K] used transposed (C = A * B^T)
// Block: 256 threads = 8 waves, wave grid 4(M) x 2(N).
// Block tile 128x64, BK=64; each wave owns a 32x32 output = 2x2 WMMA
// register tile -> 8 v_wmma_f32_16x16x32_bf16 per K-step.
// A tile is staged with CDNA5 GLOBAL_LOAD_ASYNC_TO_LDS_B128 (ASYNCcnt),
// B tile with regular b128 loads (+ LDS transpose for the NN case).

#define BM 128
#define BN 64
#define BK 64

static __device__ __forceinline__ v8f wmma_bf16(const v16bf& a, const v16bf& b, const v8f& c) {
    return __builtin_amdgcn_wmma_f32_16x16x32_bf16(
        /*neg_a=*/false, a, /*neg_b=*/false, b,
        /*c_mod=*/(short)0, c, /*reuse_a=*/false, /*reuse_b=*/false);
}

template <int TRANSB>
__global__ __launch_bounds__(256)
void k_gemm_bf16(const __bf16* __restrict__ A, const __bf16* __restrict__ B,
                 float* __restrict__ C, int M, int N, int K,
                 int lda, int ldb, int ldc) {
    __shared__ __align__(16) __bf16 As[BM][BK];   // As[m][k]   (16 KB)
    __shared__ __align__(16) __bf16 Bs[BN][BK];   // Bs[n][k]   ( 8 KB)

    int tid  = threadIdx.x;
    int lane = tid & 31;
    int wid  = tid >> 5;       // 0..7
    int wm   = wid >> 1;       // 0..3 -> 32-row band
    int wn   = wid & 1;        // 0..1 -> 32-col band
    int m0   = blockIdx.y * BM;
    int n0   = blockIdx.x * BN;
    int half = lane >> 4;      // K-half selector per ISA layout
    int l16  = lane & 15;

    v8f acc00 = {}, acc01 = {}, acc10 = {}, acc11 = {};

    for (int k0 = 0; k0 < K; k0 += BK) {
        // ---- stage A tile: 128x64 bf16 = 1024 x 16B packets, 4 per thread,
        //      via async global->LDS DMA (no VGPR round-trip, ASYNCcnt-tracked)
        #pragma unroll
        for (int p = 0; p < 4; p++) {
            int pk = tid + p * 256;
            int r  = pk >> 3;
            int kc = (pk & 7) * 8;
            unsigned ldsa = (unsigned)(uintptr_t)&As[r][kc];
            const __bf16* gp = &A[(size_t)(m0 + r) * lda + k0 + kc];
            asm volatile("global_load_async_to_lds_b128 %0, %1, off"
                         :: "v"(ldsa), "v"(gp) : "memory");
        }
        // ---- stage B tile into Bs[n][k]: 64x64 bf16 = 512 x 16B packets, 2/thread
        if (TRANSB) {
            // B is [N x K]: k-contiguous 16B load, straight 16B LDS store
            #pragma unroll
            for (int p = 0; p < 2; p++) {
                int pk = tid + p * 256;
                int n  = pk >> 3;
                int kc = (pk & 7) * 8;
                *(v8bf*)&Bs[n][kc] = *(const v8bf*)&B[(size_t)(n0 + n) * ldb + k0 + kc];
            }
        } else {
            // B is [K x N]: 16B load along N (coalesced), transpose via b16 LDS stores
            #pragma unroll
            for (int p = 0; p < 2; p++) {
                int pk = tid + p * 256;
                int kk = pk >> 3;
                int nc = (pk & 7) * 8;
                v8bf t = *(const v8bf*)&B[(size_t)(k0 + kk) * ldb + n0 + nc];
                #pragma unroll
                for (int i = 0; i < 8; i++) Bs[nc + i][kk] = t[i];
            }
        }
        // ---- uniform next-tile prefetch (single branch per iteration)
        if (k0 + BK < K) {
            int r = tid >> 3, kc = (tid & 7) * 8;
            __builtin_prefetch(&A[(size_t)(m0 + r) * lda + k0 + BK + kc], 0, 1);
            if (TRANSB)
                __builtin_prefetch(&B[(size_t)(n0 + (tid >> 3)) * ldb + k0 + BK + kc], 0, 1);
            else
                __builtin_prefetch(&B[(size_t)(k0 + BK + (tid >> 3)) * ldb + n0 + (tid & 7) * 8], 0, 1);
        }
        asm volatile("s_wait_asynccnt 0x0" ::: "memory");
        __syncthreads();

        // ---- 2 x (2x2 WMMA) over the two 32-wide K slices of the staged tile
        #pragma unroll
        for (int ks = 0; ks < BK; ks += 32) {
            v16bf av0, av1, bv0, bv1;
            int ar0 = wm * 32 + l16;
            int ar1 = wm * 32 + 16 + l16;
            #pragma unroll
            for (int j = 0; j < 16; j++) {
                int kk = ks + half * 8 + j + ((j >= 8) ? 8 : 0);  // {0-7,16-23}/{8-15,24-31}
                av0[j] = As[ar0][kk];
                av1[j] = As[ar1][kk];
            }
            int bc0 = wn * 32 + l16;
            int bc1 = wn * 32 + 16 + l16;
            #pragma unroll
            for (int j = 0; j < 16; j++) {
                bv0[j] = Bs[bc0][ks + half * 8 + j + ((j >= 8) ? 8 : 0)];
                bv1[j] = Bs[bc1][ks + half * 8 + j + ((j >= 8) ? 8 : 0)];
            }
            acc00 = wmma_bf16(av0, bv0, acc00);
            acc01 = wmma_bf16(av0, bv1, acc01);
            acc10 = wmma_bf16(av1, bv0, acc10);
            acc11 = wmma_bf16(av1, bv1, acc11);
        }
        __syncthreads();
    }

    // ---- store C: VGPR i -> M = i (lanes 0-15) / i+8 (lanes 16-31); N = lane%16
    int colb = n0 + wn * 32 + l16;
    int rowb = m0 + wm * 32 + half * 8;
    #pragma unroll
    for (int i = 0; i < 8; i++) {
        C[(size_t)(rowb + i)      * ldc + colb]      = acc00[i];
        C[(size_t)(rowb + i)      * ldc + colb + 16] = acc01[i];
        C[(size_t)(rowb + 16 + i) * ldc + colb]      = acc10[i];
        C[(size_t)(rowb + 16 + i) * ldc + colb + 16] = acc11[i];
    }
}

// ---------------------------------------------------------------- launch

extern "C" void kernel_launch(void* const* d_in, const int* in_sizes, int n_in,
                              void* d_out, int out_size, void* d_ws, size_t ws_size,
                              hipStream_t stream) {
    (void)in_sizes; (void)n_in; (void)out_size; (void)ws_size;

    const float* image_feats = (const float*)d_in[0];   // [N_IMG, IMG_DIM]
    const float* attributes  = (const float*)d_in[1];   // [N_CLS, ATT_DIM]
    const int*   labels      = (const int*)d_in[2];     // [N_IMG]
    const int*   tpl         = (const int*)d_in[3];     // [N_CLS]
    const float* att_g_a     = (const float*)d_in[4];   // [ATT_DIM, ATT_HC]
    const float* att_g_v     = (const float*)d_in[5];   // [ATT_DIM, ATT_HC]
    float*       out         = (float*)d_out;           // [N_CLS, IMG_DIM]

    // Workspace carve-out (256-byte aligned slices).
    char* ws = (char*)d_ws;
    size_t off = 0;
    auto carve = [&](size_t bytes) -> char* {
        char* p = ws + off;
        off += (bytes + 255) & ~(size_t)255;
        return p;
    };
    __bf16* attr_bf     = (__bf16*)carve((size_t)N_CLS * ATT_DIM * 2);
    __bf16* ga_bf       = (__bf16*)carve((size_t)ATT_DIM * ATT_HC * 2);
    __bf16* gv_bf       = (__bf16*)carve((size_t)ATT_DIM * ATT_HC * 2);
    float*  sums        = (float*) carve((size_t)N_CLS * IMG_DIM * 4);
    float*  counts      = (float*) carve((size_t)N_CLS * 4);
    __bf16* protos_bf   = (__bf16*)carve((size_t)N_CLS * IMG_DIM * 2);
    float*  a_f32       = (float*) carve((size_t)N_CLS * ATT_HC * 4);   // a1 / a2
    __bf16* an_bf       = (__bf16*)carve((size_t)N_CLS * ATT_HC * 2);   // a1n / a2n
    float*  d_f32       = (float*) carve((size_t)N_CLS * N_CLS * 4);    // d1 / d2
    __bf16* attn_bf     = (__bf16*)carve((size_t)N_CLS * N_CLS * 2);    // attn1 / attention
    float*  att_outs    = (float*) carve((size_t)N_CLS * ATT_DIM * 4);
    __bf16* att_outs_bf = (__bf16*)carve((size_t)N_CLS * ATT_DIM * 2);

    const int T256 = 256;
    auto gsz = [](size_t n) { return (unsigned)((n + 255) / 256); };

    // 0) operand conversions (4-wide)
    k_f32_to_bf16_x4<<<gsz((size_t)N_CLS * ATT_DIM / 4), T256, 0, stream>>>(attributes, attr_bf, N_CLS * ATT_DIM / 4);
    k_f32_to_bf16_x4<<<gsz((size_t)ATT_DIM * ATT_HC / 4), T256, 0, stream>>>(att_g_a, ga_bf, ATT_DIM * ATT_HC / 4);
    k_f32_to_bf16_x4<<<gsz((size_t)ATT_DIM * ATT_HC / 4), T256, 0, stream>>>(att_g_v, gv_bf, ATT_DIM * ATT_HC / 4);

    // 1) segment mean -> protos (bandwidth stage: one 512 MB streaming pass)
    k_zero_f32<<<gsz((size_t)N_CLS * IMG_DIM), T256, 0, stream>>>(sums, N_CLS * IMG_DIM);
    k_zero_f32<<<gsz(N_CLS), T256, 0, stream>>>(counts, N_CLS);
    k_scatter_add<<<N_IMG, T256, 0, stream>>>(image_feats, labels, sums, counts);
    k_protos<<<N_CLS, T256, 0, stream>>>(sums, counts, tpl, protos_bf);

    dim3 blk(256);
    // 2) a1 = attributes @ att_g_a            [2048x512]
    k_gemm_bf16<0><<<dim3(ATT_HC / BN, N_CLS / BM), blk, 0, stream>>>(
        attr_bf, ga_bf, a_f32, N_CLS, ATT_HC, ATT_DIM, ATT_DIM, ATT_HC, ATT_HC);
    // 3) a1n (bf16)
    k_rownorm_bf16<<<N_CLS, T256, 0, stream>>>(a_f32, an_bf, ATT_HC);
    // 4) d1 = a1n @ a1n^T                     [2048x2048]
    k_gemm_bf16<1><<<dim3(N_CLS / BN, N_CLS / BM), blk, 0, stream>>>(
        an_bf, an_bf, d_f32, N_CLS, N_CLS, ATT_HC, ATT_HC, ATT_HC, N_CLS);
    // 5) attn1 = softmax(mask(d1)*T)          (bf16 weights)
    k_mask_softmax_bf16<<<N_CLS, T256, 0, stream>>>(d_f32, attn_bf, N_CLS);
    // 6) att_outs = attn1 @ attributes        [2048x1024]
    k_gemm_bf16<0><<<dim3(ATT_DIM / BN, N_CLS / BM), blk, 0, stream>>>(
        attn_bf, attr_bf, att_outs, N_CLS, ATT_DIM, N_CLS, N_CLS, ATT_DIM, ATT_DIM);
    k_f32_to_bf16_x4<<<gsz((size_t)N_CLS * ATT_DIM / 4), T256, 0, stream>>>(att_outs, att_outs_bf, N_CLS * ATT_DIM / 4);
    // 7) a2 = att_outs @ att_g_v              [2048x512]
    k_gemm_bf16<0><<<dim3(ATT_HC / BN, N_CLS / BM), blk, 0, stream>>>(
        att_outs_bf, gv_bf, a_f32, N_CLS, ATT_HC, ATT_DIM, ATT_DIM, ATT_HC, ATT_HC);
    // 8) a2n (bf16)
    k_rownorm_bf16<<<N_CLS, T256, 0, stream>>>(a_f32, an_bf, ATT_HC);
    // 9) d2 = a2n @ a2n^T
    k_gemm_bf16<1><<<dim3(N_CLS / BN, N_CLS / BM), blk, 0, stream>>>(
        an_bf, an_bf, d_f32, N_CLS, N_CLS, ATT_HC, ATT_HC, ATT_HC, N_CLS);
    // 10) attention = softmax(mask(d2)*T)
    k_mask_softmax_bf16<<<N_CLS, T256, 0, stream>>>(d_f32, attn_bf, N_CLS);
    // 11) out = attention @ protos            [2048x2048] -> d_out
    k_gemm_bf16<0><<<dim3(IMG_DIM / BN, N_CLS / BM), blk, 0, stream>>>(
        attn_bf, protos_bf, out, N_CLS, IMG_DIM, N_CLS, N_CLS, IMG_DIM, IMG_DIM);
}
